// APPNP_Net_86706799772257
// MI455X (gfx1250) — matrix operand order repaired and checked by reading, not compiled, not run on hardware
//
#include <hip/hip_runtime.h>
#include <hip/hip_bf16.h>

typedef __attribute__((ext_vector_type(16))) _Float16 v16h;
typedef __attribute__((ext_vector_type(8)))  _Float16 v8h;
typedef __attribute__((ext_vector_type(8)))  float    v8f;

#define NNODES 100000
#define NEDGES 3200000
#define FIN    256
#define FHID   256
#define FOUT   64
#define KSTEPS 3
#define ALPHA  0.1f

#define WAVES_PER_BLOCK 2                 // 64 threads; 6250 row-tiles / 2 = 3125 blocks exact
#define ROWTILES (NNODES / 16)            // 6250
#define NCHUNKS  (FIN / 32)               // 8 K-chunks
#define CHUNK_HALFS (FHID * 32)           // 8192 f16 = 16 KB per W1 K-chunk

// ---------------------------------------------------------------------------
// CDNA5 async global->LDS copy (ASYNCcnt-tracked), inline asm (toolchain-portable)
// ---------------------------------------------------------------------------
__device__ __forceinline__ void async_ld_b128(unsigned lds_off, const void* gaddr) {
    asm volatile("global_load_async_to_lds_b128 %0, %1, off"
                 :: "v"(lds_off), "v"(gaddr) : "memory");
}
#define ASYNC_WAIT_16() asm volatile("s_wait_asynccnt 0x10" ::: "memory")
#define ASYNC_WAIT_0()  asm volatile("s_wait_asynccnt 0x0"  ::: "memory")

// ---------------------------------------------------------------------------
// Kernel 0: transpose+convert weights to column-major f16
//   W1t[col*FIN + k] = W1[k*FHID + col]   (256 cols x 256 k)
//   W2t[col*FHID + k] = W2[k*FOUT + col]  (64 cols x 256 k)
// ---------------------------------------------------------------------------
__global__ void convert_weights(const float* __restrict__ W1,
                                const float* __restrict__ W2,
                                _Float16* __restrict__ W1t,
                                _Float16* __restrict__ W2t) {
    int idx = blockIdx.x * blockDim.x + threadIdx.x;
    if (idx < FIN * FHID) {
        int col = idx / FIN;
        int k   = idx % FIN;
        W1t[idx] = (_Float16)W1[k * FHID + col];
    }
    if (idx < FHID * FOUT) {
        int col = idx / FHID;
        int k   = idx % FHID;
        W2t[idx] = (_Float16)W2[k * FOUT + col];
    }
}

// ---------------------------------------------------------------------------
// Degree / normalization (GCN norm with self-loops)
// ---------------------------------------------------------------------------
__global__ void deg_init(float* __restrict__ deg) {
    int i = blockIdx.x * blockDim.x + threadIdx.x;
    if (i < NNODES) deg[i] = 1.0f;   // self-loop contribution
}

__global__ void deg_count(const int* __restrict__ dst, float* __restrict__ deg) {
    int e = blockIdx.x * blockDim.x + threadIdx.x;
    if (e < NEDGES) atomicAdd(&deg[dst[e]], 1.0f);
}

__global__ void dinv_kernel(const float* __restrict__ deg, float* __restrict__ dinv) {
    int i = blockIdx.x * blockDim.x + threadIdx.x;
    if (i < NNODES) {
        float d = deg[i];
        dinv[i] = (d > 0.0f) ? __frsqrt_rn(d) : 0.0f;
    }
}

// ---------------------------------------------------------------------------
// Fused MLP: h = relu(x@W1 + b1) @ W2 + b2  via v_wmma_f32_16x16x32_f16.
// One wave owns a 16-row strip. W1/W2 fragments staged in LDS with async
// double-buffered copies so WMMAs never wait on global loads.
// ---------------------------------------------------------------------------
__global__ __launch_bounds__(WAVES_PER_BLOCK * 32)
void mlp_kernel(const float* __restrict__ x,
                const _Float16* __restrict__ W1t, const float* __restrict__ b1,
                const _Float16* __restrict__ W2t, const float* __restrict__ b2,
                float* __restrict__ h) {
    __shared__ _Float16 sB[2 * CHUNK_HALFS];                // 32 KB: W1 chunks / whole W2
    __shared__ _Float16 sHid[WAVES_PER_BLOCK][16 * FHID];   // 16 KB: hidden staging

    const int tid  = threadIdx.x;
    const int wave = tid >> 5;
    const int lane = tid & 31;
    const int tile = blockIdx.x * WAVES_PER_BLOCK + wave;   // always < ROWTILES (exact grid)
    const int row0 = tile * 16;
    const int m    = lane & 15;   // row (A) / column (B, C) within a 16-tile
    const int kg   = lane >> 4;   // K-half selector

    const unsigned sB_base = (unsigned)(size_t)&sB[0];

    // Stage a 32-K-wide chunk of W1t into LDS buffer `buf` (block-cooperative).
    // 1024 b128 transfers / 64 threads = 16 per thread = 16 async ops per wave.
    auto stage_w1_chunk = [&](int c, int buf) {
        unsigned base = sB_base + (unsigned)(buf * CHUNK_HALFS) * 2u;
#pragma unroll
        for (int j = 0; j < 16; ++j) {
            int idx = tid * 16 + j;        // b128 index in chunk
            int col = idx >> 2;            // 4 x b128 (32 halfs) per column
            int kk  = (idx & 3) * 8;
            async_ld_b128(base + (unsigned)(col * 32 + kk) * 2u,
                          W1t + (size_t)col * FIN + c * 32 + kk);
        }
    };

    // ---------------- GEMM1: hidden = relu(x @ W1 + b1) -------------------
    v8f acc1[16];
#pragma unroll
    for (int t = 0; t < 16; ++t) acc1[t] = (v8f){0,0,0,0,0,0,0,0};

    const float* xrow = x + (size_t)(row0 + m) * FIN;

    stage_w1_chunk(0, 0);
    stage_w1_chunk(1, 1);

    for (int c = 0; c < NCHUNKS; ++c) {
        if (c < NCHUNKS - 1) { ASYNC_WAIT_16(); } else { ASYNC_WAIT_0(); }
        __syncthreads();                       // chunk c visible to both waves

        const int ks = c * 32;
        __builtin_prefetch(xrow + ks + 64, 0, 0);

        // A fragment: halves 0..15 = K{ks+kg*16..ks+kg*16+15}
        v8f xa = *(const v8f*)(xrow + ks + kg * 16);
        v8f xb = *(const v8f*)(xrow + ks + kg * 16 + 8);
        v16h a;
#pragma unroll
        for (int j = 0; j < 8; ++j) { a[j] = (_Float16)xa[j]; a[8 + j] = (_Float16)xb[j]; }

        const _Float16* sBc = sB + (c & 1) * CHUNK_HALFS;
#pragma unroll
        for (int t = 0; t < 16; ++t) {
            v16h bfrag = *(const v16h*)(sBc + (t * 16 + m) * 32 + kg * 16);
            acc1[t] = __builtin_amdgcn_wmma_f32_16x16x32_f16(
                false, a, false, bfrag, (short)0, acc1[t], false, false);
        }

        __syncthreads();                       // everyone done reading buf (c&1)
        if (c + 2 < NCHUNKS) stage_w1_chunk(c + 2, c & 1);
    }

    // Stage the whole W2t (32 KB) into the now-free double buffer (linear copy).
#pragma unroll
    for (int j = 0; j < 32; ++j) {
        int idx = tid * 32 + j;                // 2048 x b128
        async_ld_b128(sB_base + (unsigned)idx * 16u,
                      (const char*)W2t + (size_t)idx * 16);
    }

    // bias + ReLU, stage hidden tile row-major f16 in per-wave LDS
    _Float16* lh = sHid[wave];
#pragma unroll
    for (int t = 0; t < 16; ++t) {
        float bias = b1[t * 16 + m];
#pragma unroll
        for (int r = 0; r < 8; ++r) {
            float v = acc1[t][r] + bias;
            v = v > 0.0f ? v : 0.0f;
            lh[(r + kg * 8) * FHID + t * 16 + m] = (_Float16)v;   // (row, col)
        }
    }

    ASYNC_WAIT_0();
    __syncthreads();                           // W2 staged and visible

    // ---------------- GEMM2: out = hidden @ W2 + b2 ------------------------
    v8f acc2[4];
#pragma unroll
    for (int t = 0; t < 4; ++t) acc2[t] = (v8f){0,0,0,0,0,0,0,0};

    const _Float16* sw2 = sB;                  // W2t image: [col*FHID + k]
    for (int ks = 0; ks < FHID; ks += 32) {
        const _Float16* ap = lh + m * FHID + ks + kg * 16;
        v8h a0 = *(const v8h*)ap;
        v8h a1 = *(const v8h*)(ap + 8);
        v16h a;
#pragma unroll
        for (int j = 0; j < 8; ++j) { a[j] = a0[j]; a[8 + j] = a1[j]; }

#pragma unroll
        for (int t = 0; t < 4; ++t) {
            v16h bfrag = *(const v16h*)(sw2 + (size_t)(t * 16 + m) * FHID + ks + kg * 16);
            acc2[t] = __builtin_amdgcn_wmma_f32_16x16x32_f16(
                false, a, false, bfrag, (short)0, acc2[t], false, false);
        }
    }

    // epilogue: C/D layout -> row = r + kg*8, col = t*16 + m
#pragma unroll
    for (int t = 0; t < 4; ++t) {
        float bias = b2[t * 16 + m];
#pragma unroll
        for (int r = 0; r < 8; ++r) {
            h[(size_t)(row0 + r + kg * 8) * FOUT + t * 16 + m] = acc2[t][r] + bias;
        }
    }
}

// ---------------------------------------------------------------------------
// Propagation kernels
// ---------------------------------------------------------------------------
// agg[i] = dinv[i]^2 * cur[i]   (self-loop term, also initializes agg)
__global__ void init_agg(float* __restrict__ agg, const float* __restrict__ dinv,
                         const float* __restrict__ cur) {
    int idx = blockIdx.x * blockDim.x + threadIdx.x;
    if (idx < NNODES * FOUT) {
        int i = idx >> 6;
        float di = dinv[i];
        agg[idx] = di * di * cur[idx];
    }
}

// 16 lanes per edge: float4 gather of cur[src] + 4 atomic f32 scatter-adds
__global__ void edge_scatter(const int* __restrict__ src, const int* __restrict__ dst,
                             const float* __restrict__ dinv,
                             const float* __restrict__ cur, float* __restrict__ agg) {
    size_t gid = (size_t)blockIdx.x * blockDim.x + threadIdx.x;
    int e = (int)(gid >> 4);
    int g = (int)(gid & 15);
    if (e >= NEDGES) return;
    int s = src[e];
    int d = dst[e];
    float nrm = dinv[s] * dinv[d];
    const float4 v = *(const float4*)(cur + (size_t)s * FOUT + g * 4);
    float* base = agg + (size_t)d * FOUT + g * 4;
    atomicAdd(base + 0, nrm * v.x);
    atomicAdd(base + 1, nrm * v.y);
    atomicAdd(base + 2, nrm * v.z);
    atomicAdd(base + 3, nrm * v.w);
}

// nxt = (1-ALPHA)*agg + ALPHA*h
__global__ void combine(float* __restrict__ nxt, const float* __restrict__ agg,
                        const float* __restrict__ h) {
    int idx = blockIdx.x * blockDim.x + threadIdx.x;
    if (idx < NNODES * FOUT) {
        nxt[idx] = (1.0f - ALPHA) * agg[idx] + ALPHA * h[idx];
    }
}

// ---------------------------------------------------------------------------
// Launch
// ---------------------------------------------------------------------------
static inline size_t align256(size_t v) { return (v + 255) & ~(size_t)255; }

extern "C" void kernel_launch(void* const* d_in, const int* in_sizes, int n_in,
                              void* d_out, int out_size, void* d_ws, size_t ws_size,
                              hipStream_t stream) {
    const float* x   = (const float*)d_in[0];
    const int*   ei  = (const int*)d_in[1];   // [2, E]: row0 = src, row1 = dst
    const float* W1  = (const float*)d_in[2];
    const float* b1  = (const float*)d_in[3];
    const float* W2  = (const float*)d_in[4];
    const float* b2  = (const float*)d_in[5];
    float*       out = (float*)d_out;

    const int* e_src = ei;
    const int* e_dst = ei + NEDGES;

    // workspace carve-up
    char* p = (char*)d_ws;
    _Float16* W1t = (_Float16*)p; p += align256((size_t)FIN * FHID * 2);
    _Float16* W2t = (_Float16*)p; p += align256((size_t)FHID * FOUT * 2);
    float* deg    = (float*)p;    p += align256((size_t)NNODES * 4);
    float* dinv   = (float*)p;    p += align256((size_t)NNODES * 4);
    float* hbuf   = (float*)p;    p += align256((size_t)NNODES * FOUT * 4);
    float* aggbuf = (float*)p;    p += align256((size_t)NNODES * FOUT * 4);
    float* curbuf = (float*)p;    p += align256((size_t)NNODES * FOUT * 4);

    // 1) weights -> col-major f16
    convert_weights<<<(FIN * FHID + 255) / 256, 256, 0, stream>>>(W1, W2, W1t, W2t);

    // 2) degrees + dinv
    deg_init<<<(NNODES + 255) / 256, 256, 0, stream>>>(deg);
    deg_count<<<(NEDGES + 255) / 256, 256, 0, stream>>>(e_dst, deg);
    dinv_kernel<<<(NNODES + 255) / 256, 256, 0, stream>>>(deg, dinv);

    // 3) fused WMMA MLP -> hbuf  (3125 blocks x 2 waves = 6250 tiles exact)
    mlp_kernel<<<ROWTILES / WAVES_PER_BLOCK, WAVES_PER_BLOCK * 32, 0, stream>>>(
        x, W1t, b1, W2t, b2, hbuf);

    // 4) APPNP propagation, K = 3
    const int elem_blocks = (NNODES * FOUT + 255) / 256;
    const size_t scat_threads = (size_t)NEDGES * 16;
    const int scat_blocks = (int)((scat_threads + 255) / 256);

    const float* cur = hbuf;
    for (int step = 0; step < KSTEPS; ++step) {
        init_agg<<<elem_blocks, 256, 0, stream>>>(aggbuf, dinv, cur);
        edge_scatter<<<scat_blocks, 256, 0, stream>>>(e_src, e_dst, dinv, cur, aggbuf);
        float* nxt = (step == KSTEPS - 1) ? out : curbuf;
        combine<<<elem_blocks, 256, 0, stream>>>(nxt, aggbuf, hbuf);
        cur = curbuf;
    }
}
